// GraphResoningModule_66374424592665
// MI455X (gfx1250) — matrix operand order repaired and checked by reading, not compiled, not run on hardware
//
#include <hip/hip_runtime.h>
#include <stdint.h>

// ---------------------------------------------------------------------------
// CDNA5 (gfx1250) wave32 WMMA pipeline for the graph-reasoning module.
// All heavy GEMMs use v_wmma_f32_16x16x32_bf16 with f32 accumulation.
// Cosine-similarity bound (|cos|<=1) lets us do single-pass exp + rowsum
// softmax with no max subtraction and no online rescaling.
// K-loops are capped at unroll 2: enough load/WMMA overlap for the
// scheduler, without hoisting every A-fragment into registers (which
// caused scratch spills in the previous revision).
// ---------------------------------------------------------------------------

typedef __attribute__((ext_vector_type(16))) __bf16 v16bf;
typedef __attribute__((ext_vector_type(8)))  float  v8f;

#define B_  8
#define D_  512
#define HW_ 4096
#define IT_ 256
#define OD_ 512

__device__ __forceinline__ uint16_t f2bf(float x) {
  uint32_t u = __float_as_uint(x);
  u += 0x7FFFu + ((u >> 16) & 1u);          // round-to-nearest-even
  return (uint16_t)(u >> 16);
}

union ABf { uint32_t u[8]; v16bf v; };

// A-fragment (16x32 bf16, M across lanes 0-15, K in VGPRs per ISA 7.12.2).
// rowp = this lane's row pointer (row = lane&15).  The 8 u32 loads form two
// contiguous 16B blocks at kk+8*hi and kk+16+8*hi -> lower to 2x b128.
__device__ __forceinline__ void loadA(ABf& a, const uint16_t* rowp, int kk, int hi) {
#pragma unroll
  for (int vi = 0; vi < 8; ++vi) {
    int kb = ((vi & 4) << 2) + hi * 8 + ((vi & 3) << 1);   // 16*(vi>=4)+8*hi+2*(vi&3)
    a.u[vi] = *(const uint32_t*)(rowp + kk + kb);
  }
}

// B-fragment (32x16 bf16, N across lanes, K in VGPRs).  For X*Y^T the lane's
// B column equals a row of Y: two 16B blocks at kk+8*hi and kk+16+8*hi.
__device__ __forceinline__ void loadB(ABf& b, const uint16_t* rowp, int kk, int hi) {
  const uint4* p = (const uint4*)(rowp + kk + hi * 8);
  uint4 q0 = p[0], q1 = p[2];
  b.u[0] = q0.x; b.u[1] = q0.y; b.u[2] = q0.z; b.u[3] = q0.w;
  b.u[4] = q1.x; b.u[5] = q1.y; b.u[6] = q1.z; b.u[7] = q1.w;
}

__device__ __forceinline__ v8f wmma_bf16(const ABf& a, const ABf& b, v8f c) {
  return __builtin_amdgcn_wmma_f32_16x16x32_bf16(false, a.v, false, b.v,
                                                 (short)0, c, false, false);
}

// ---------------------------------------------------------------------------
// Kernel 1a: f32 -> bf16 bulk convert (feat_map, phi_w, theta_w).
// ---------------------------------------------------------------------------
__global__ void convert_bf16(const float* __restrict__ in,
                             uint16_t* __restrict__ out, int n) {
  int i = (blockIdx.x * blockDim.x + threadIdx.x) * 4;
  if (i + 3 < n) {
    float4 v = *(const float4*)(in + i);
    uint32_t p0 = (uint32_t)f2bf(v.x) | ((uint32_t)f2bf(v.y) << 16);
    uint32_t p1 = (uint32_t)f2bf(v.z) | ((uint32_t)f2bf(v.w) << 16);
    *(uint32_t*)(out + i)     = p0;
    *(uint32_t*)(out + i + 2) = p1;
  }
}

// Kernel 1b: weight [c][d] f32 -> wT [d][c] bf16 (K-contiguous B operand).
__global__ void transpose_w_bf16(const float* __restrict__ w,
                                 uint16_t* __restrict__ wT) {
  int idx = blockIdx.x * blockDim.x + threadIdx.x;   // = d*512 + c
  int d = idx >> 9, c = idx & 511;
  wT[idx] = f2bf(w[c * OD_ + d]);
}

// ---------------------------------------------------------------------------
// Kernel 2: phin/thetan = row-normalized (feat_vec @ {phi,theta}_w^T), bf16.
// One WG = one batch b + 16-row block of n.  feat tile transposed via LDS.
// waves 0-3: phi, waves 4-7: theta;   wave does 4 N-tiles, K=512 loop.
// ---------------------------------------------------------------------------
__global__ void phitheta_kernel(const uint16_t* __restrict__ featc,
                                const uint16_t* __restrict__ phw,
                                const uint16_t* __restrict__ thw,
                                uint16_t* __restrict__ phin,
                                uint16_t* __restrict__ thetan) {
  __shared__ uint16_t At[16][520];     // feat tile transposed: [n][c]
  __shared__ float    Ph[16][260];     // unnormalized phi rows
  __shared__ float    Th[16][260];     // unnormalized theta rows
  __shared__ float    nrm[2][16];

  int wg = blockIdx.x;
  int b  = wg >> 8;                    // HW/16 = 256 row-blocks per batch
  int n0 = (wg & 255) * 16;
  int t  = threadIdx.x;

  // Stage feat[c][n0..n0+15] -> At[n][c]  (coalesced 32B row segments)
  {
    int nl = (t & 7) * 2;
    const uint16_t* fp = featc + (size_t)b * D_ * HW_ + n0;
#pragma unroll 4
    for (int it = 0; it < 16; ++it) {
      int c = (t >> 3) + it * 32;
      uint32_t w = *(const uint32_t*)(fp + (size_t)c * HW_ + nl);
      At[nl][c]     = (uint16_t)(w & 0xFFFF);
      At[nl + 1][c] = (uint16_t)(w >> 16);
    }
  }
  __syncthreads();

  int wave = t >> 5, lane = t & 31, col = lane & 15, hi = lane >> 4;
  int s = wave >> 2;                               // 0 = phi, 1 = theta
  const uint16_t* Wm = s ? thw : phw;
  float (*Obuf)[260] = s ? Th : Ph;
  const uint16_t* arow = &At[col][0];              // lane's A row in LDS

#pragma unroll 1
  for (int jj = 0; jj < 4; ++jj) {
    int j = (wave & 3) * 4 + jj;                   // N-tile 0..15 (i = j*16+col)
    v8f acc = {};
    const uint16_t* wrow = Wm + (size_t)(j * 16 + col) * D_;
#pragma unroll 2
    for (int kk = 0; kk < D_; kk += 32) {
      ABf a, bb;
      loadA(a, arow, kk, hi);
      loadB(bb, wrow, kk, hi);
      acc = wmma_bf16(a, bb, acc);
    }
#pragma unroll
    for (int v = 0; v < 8; ++v)
      Obuf[hi * 8 + v][j * 16 + col] = acc[v];
  }
  __syncthreads();

  // Row norms: 16 threads per row, shfl-reduce within 16-lane groups.
  {
    int row = t >> 4, part = t & 15;
    float sp = 0.f, st = 0.f;
#pragma unroll
    for (int k = 0; k < 16; ++k) {
      float x = Ph[row][part * 16 + k]; sp += x * x;
      float y = Th[row][part * 16 + k]; st += y * y;
    }
#pragma unroll
    for (int m = 1; m < 16; m <<= 1) {
      sp += __shfl_xor(sp, m, 16);
      st += __shfl_xor(st, m, 16);
    }
    if (part == 0) {
      nrm[0][row] = rsqrtf(sp + 1e-20f);
      nrm[1][row] = rsqrtf(st + 1e-20f);
    }
  }
  __syncthreads();

  // Normalize, convert to bf16, store (32B contiguous per thread).
  {
    int row = t >> 4, i0 = (t & 15) * 16;
    float ip = nrm[0][row], iq = nrm[1][row];
    size_t base = ((size_t)b * HW_ + n0 + row) * IT_ + i0;
#pragma unroll
    for (int k = 0; k < 8; ++k) {
      uint32_t pp = (uint32_t)f2bf(Ph[row][i0 + 2 * k] * ip) |
                    ((uint32_t)f2bf(Ph[row][i0 + 2 * k + 1] * ip) << 16);
      uint32_t tt = (uint32_t)f2bf(Th[row][i0 + 2 * k] * iq) |
                    ((uint32_t)f2bf(Th[row][i0 + 2 * k + 1] * iq) << 16);
      ((uint32_t*)(phin + base))[k]   = pp;
      ((uint32_t*)(thetan + base))[k] = tt;
    }
  }
}

// ---------------------------------------------------------------------------
// Kernel 3: S = exp(phin @ thetan^T) bf16  +  exact row sums (denominators).
// |cos|<=1 -> exp is stable with no max pass.  WG owns 64 rows x all 4096 m.
// wave: row-tile (w&3), m-half (w>>2)*2048; 4 N-tiles per A-frag reload.
// ---------------------------------------------------------------------------
__global__ void sim_kernel(const uint16_t* __restrict__ phin,
                           const uint16_t* __restrict__ thetan,
                           uint16_t* __restrict__ S,
                           float* __restrict__ denom) {
  __shared__ float dn[64];
  int wg = blockIdx.x;
  int b  = wg >> 6;                     // HW/64 = 64 row-blocks per batch
  int n0 = (wg & 63) * 64;
  int t  = threadIdx.x;
  if (t < 64) dn[t] = 0.f;
  __syncthreads();

  int wave = t >> 5, lane = t & 31, col = lane & 15, hi = lane >> 4;
  int mt = wave & 3;                    // which 16-row tile
  int mh = wave >> 2;                   // which 2048-col half
  const uint16_t* arow = phin + ((size_t)b * HW_ + n0 + mt * 16 + col) * IT_;

  float rp[8] = {0, 0, 0, 0, 0, 0, 0, 0};
#pragma unroll 1
  for (int jg = 0; jg < 32; ++jg) {
    int j0 = mh * 128 + jg * 4;         // first of 4 N-tiles
    const uint16_t* brow[4];
#pragma unroll
    for (int jj = 0; jj < 4; ++jj)
      brow[jj] = thetan + ((size_t)b * HW_ + (size_t)(j0 + jj) * 16 + col) * IT_;

    v8f acc[4] = {};
#pragma unroll 2
    for (int kk = 0; kk < IT_; kk += 32) {
      ABf a;
      loadA(a, arow, kk, hi);
#pragma unroll
      for (int jj = 0; jj < 4; ++jj) {
        ABf bb;
        loadB(bb, brow[jj], kk, hi);
        acc[jj] = wmma_bf16(a, bb, acc[jj]);
      }
    }
#pragma unroll
    for (int jj = 0; jj < 4; ++jj) {
      int m = (j0 + jj) * 16 + col;
#pragma unroll
      for (int v = 0; v < 8; ++v) {
        float e = __expf(acc[jj][v]);
        rp[v] += e;
        int n = n0 + mt * 16 + hi * 8 + v;
        S[((size_t)b * HW_ + n) * HW_ + m] = f2bf(e);
      }
    }
  }
  // reduce partial denominators across the 16-lane group, combine halves in LDS
#pragma unroll
  for (int v = 0; v < 8; ++v) {
#pragma unroll
    for (int m = 1; m < 16; m <<= 1) rp[v] += __shfl_xor(rp[v], m, 16);
  }
  if (col == 0) {
#pragma unroll
    for (int v = 0; v < 8; ++v) atomicAdd(&dn[mt * 16 + hi * 8 + v], rp[v]);
  }
  __syncthreads();
  if (t < 64) denom[(size_t)b * HW_ + n0 + t] = dn[t];
}

// ---------------------------------------------------------------------------
// Kernel 4: T = (S @ feat_vec) / denom, bf16.   B operand = featc[c][m] is
// already K(=m)-contiguous per lane.  WG owns 64 rows x 512 c; wave: row-tile
// (w&3), c-half (w>>2)*256; K=4096, 4 N-tiles per A-frag reload.
// ---------------------------------------------------------------------------
__global__ void pv_kernel(const uint16_t* __restrict__ S,
                          const uint16_t* __restrict__ featc,
                          const float* __restrict__ denom,
                          uint16_t* __restrict__ T) {
  int wg = blockIdx.x;
  int b  = wg >> 6;
  int n0 = (wg & 63) * 64;
  int t  = threadIdx.x, wave = t >> 5, lane = t & 31, col = lane & 15, hi = lane >> 4;
  int mt = wave & 3, ch = wave >> 2;
  const uint16_t* arow = S + ((size_t)b * HW_ + n0 + mt * 16 + col) * HW_;

  float inv[8];
#pragma unroll
  for (int v = 0; v < 8; ++v)
    inv[v] = 1.f / denom[(size_t)b * HW_ + n0 + mt * 16 + hi * 8 + v];

#pragma unroll 1
  for (int jg = 0; jg < 4; ++jg) {
    const uint16_t* brow[4];
    int cbase[4];
#pragma unroll
    for (int jj = 0; jj < 4; ++jj) {
      int c = (ch * 16 + jg * 4 + jj) * 16 + col;
      cbase[jj] = c;
      brow[jj] = featc + ((size_t)b * D_ + c) * HW_;
    }
    v8f acc[4] = {};
#pragma unroll 2
    for (int kk = 0; kk < HW_; kk += 32) {
      ABf a;
      loadA(a, arow, kk, hi);
#pragma unroll
      for (int jj = 0; jj < 4; ++jj) {
        ABf bb;
        loadB(bb, brow[jj], kk, hi);
        acc[jj] = wmma_bf16(a, bb, acc[jj]);
      }
    }
#pragma unroll
    for (int jj = 0; jj < 4; ++jj) {
#pragma unroll
      for (int v = 0; v < 8; ++v) {
        int n = n0 + mt * 16 + hi * 8 + v;
        T[((size_t)b * HW_ + n) * D_ + cbase[jj]] = f2bf(acc[jj][v] * inv[v]);
      }
    }
  }
}

// ---------------------------------------------------------------------------
// Kernel 5: out[b][d][n] = relu(T @ weight).  B operand = wT[d][c] is
// K-contiguous.  C-tile rows are contiguous in n -> two float4 stores/lane.
// ---------------------------------------------------------------------------
__global__ void out_kernel(const uint16_t* __restrict__ T,
                           const uint16_t* __restrict__ wT,
                           float* __restrict__ out) {
  int wg = blockIdx.x;
  int b  = wg >> 6;
  int n0 = (wg & 63) * 64;
  int t  = threadIdx.x, wave = t >> 5, lane = t & 31, col = lane & 15, hi = lane >> 4;
  int mt = wave & 3, dh = wave >> 2;
  const uint16_t* arow = T + ((size_t)b * HW_ + n0 + mt * 16 + col) * D_;

#pragma unroll 1
  for (int jg = 0; jg < 4; ++jg) {
    const uint16_t* brow[4];
    int dbase[4];
#pragma unroll
    for (int jj = 0; jj < 4; ++jj) {
      int d = (dh * 16 + jg * 4 + jj) * 16 + col;
      dbase[jj] = d;
      brow[jj] = wT + (size_t)d * D_;
    }
    v8f acc[4] = {};
#pragma unroll 2
    for (int kk = 0; kk < D_; kk += 32) {
      ABf a;
      loadA(a, arow, kk, hi);
#pragma unroll
      for (int jj = 0; jj < 4; ++jj) {
        ABf bb;
        loadB(bb, brow[jj], kk, hi);
        acc[jj] = wmma_bf16(a, bb, acc[jj]);
      }
    }
#pragma unroll
    for (int jj = 0; jj < 4; ++jj) {
      float4 lo = make_float4(fmaxf(acc[jj][0], 0.f), fmaxf(acc[jj][1], 0.f),
                              fmaxf(acc[jj][2], 0.f), fmaxf(acc[jj][3], 0.f));
      float4 hi4 = make_float4(fmaxf(acc[jj][4], 0.f), fmaxf(acc[jj][5], 0.f),
                               fmaxf(acc[jj][6], 0.f), fmaxf(acc[jj][7], 0.f));
      float* op = out + ((size_t)b * OD_ + dbase[jj]) * HW_ + n0 + mt * 16 + hi * 8;
      *(float4*)op = lo;
      *(float4*)(op + 4) = hi4;
    }
  }
}

// ---------------------------------------------------------------------------
extern "C" void kernel_launch(void* const* d_in, const int* in_sizes, int n_in,
                              void* d_out, int out_size, void* d_ws, size_t ws_size,
                              hipStream_t stream) {
  const float* feat    = (const float*)d_in[0];
  const float* phi_w   = (const float*)d_in[1];
  const float* theta_w = (const float*)d_in[2];
  const float* weight  = (const float*)d_in[3];

  char* ws = (char*)d_ws;
  size_t off = 0;
  auto take = [&](size_t bytes) -> char* {
    char* p = ws + off;
    off += (bytes + 255) & ~(size_t)255;
    return p;
  };
  uint16_t* featc  = (uint16_t*)take((size_t)B_ * D_ * HW_ * 2);
  uint16_t* phw    = (uint16_t*)take((size_t)IT_ * D_ * 2);
  uint16_t* thw    = (uint16_t*)take((size_t)IT_ * D_ * 2);
  uint16_t* wT     = (uint16_t*)take((size_t)D_ * OD_ * 2);
  uint16_t* phin   = (uint16_t*)take((size_t)B_ * HW_ * IT_ * 2);
  uint16_t* thetan = (uint16_t*)take((size_t)B_ * HW_ * IT_ * 2);
  uint16_t* S      = (uint16_t*)take((size_t)B_ * HW_ * HW_ * 2);
  float*    denom  = (float*)   take((size_t)B_ * HW_ * 4);
  uint16_t* T      = (uint16_t*)take((size_t)B_ * HW_ * D_ * 2);

  convert_bf16<<<(B_ * D_ * HW_) / 1024, 256, 0, stream>>>(feat, featc, B_ * D_ * HW_);
  convert_bf16<<<(IT_ * D_) / 1024, 256, 0, stream>>>(phi_w, phw, IT_ * D_);
  convert_bf16<<<(IT_ * D_) / 1024, 256, 0, stream>>>(theta_w, thw, IT_ * D_);
  transpose_w_bf16<<<(D_ * OD_) / 256, 256, 0, stream>>>(weight, wT);

  phitheta_kernel<<<B_ * (HW_ / 16), 256, 0, stream>>>(featc, phw, thw, phin, thetan);
  sim_kernel<<<B_ * (HW_ / 64), 256, 0, stream>>>(phin, thetan, S, denom);
  pv_kernel<<<B_ * (HW_ / 64), 256, 0, stream>>>(S, featc, denom, T);
  out_kernel<<<B_ * (HW_ / 64), 256, 0, stream>>>(T, wT, (float*)d_out);
}
